// PowerSpectrumModel_34333968564629
// MI455X (gfx1250) — matrix operand order repaired and checked by reading, not compiled, not run on hardware
//
#include <hip/hip_runtime.h>

typedef __attribute__((ext_vector_type(16))) __bf16       v16bf;
typedef __attribute__((ext_vector_type(8)))  float        v8f;
typedef __attribute__((ext_vector_type(4)))  float        vf4;
typedef __attribute__((ext_vector_type(4)))  unsigned int vu4;

#define N_FEAT  1024
#define H1_DIM  256
#define H2_DIM  256
#define TILE_M  64
#define PITCH_A 34    // halfwords; 17-dword row stride -> conflict-free LDS gathers
#define PITCH_H 258   // halfwords; 129-dword row stride -> conflict-free
#define KSTEPS1 (N_FEAT / 32)   // 32
#define KSTEPS2 (H1_DIM / 32)   // 8

__device__ __forceinline__ unsigned short f2bf(float x) {
  unsigned int u = __float_as_uint(x);
  u += 0x7FFFu + ((u >> 16) & 1u);          // round-to-nearest-even
  return (unsigned short)(u >> 16);
}

__device__ __forceinline__ float silu(float x) {
  return x / (1.0f + __expf(-x));
}

union AFrag {
  v16bf v;
  unsigned int u[8];
  vu4 q[2];
};

// Gather a 16x32 bf16 A-fragment row from LDS per ISA layout:
// lanes 0-15: M=lane, V0..3 hold K=0..7, V4..7 hold K=16..23
// lanes 16-31: M=lane-16, V0..3 hold K=8..15, V4..7 hold K=24..31
__device__ __forceinline__ v16bf gather_a(const unsigned short* row, int half) {
  AFrag r;
#pragma unroll
  for (int p = 0; p < 8; ++p) {
    int kb = (p >> 2) * 16 + half * 8 + (p & 3) * 2;   // even -> dword aligned
    r.u[p] = *reinterpret_cast<const unsigned int*>(row + kb);
  }
  return r.v;
}

// Pre-swizzled B fragment: 32 contiguous bytes per lane, 1 KB per fragment.
__device__ __forceinline__ v16bf load_b(const unsigned short* frags, int frag, int lane) {
  AFrag r;
  const vu4* p = reinterpret_cast<const vu4*>(frags + (size_t)frag * 512 + lane * 16);
  r.q[0] = p[0];
  r.q[1] = p[1];
  return r.v;
}

__global__ void zero_out_kernel(float* __restrict__ out, int n) {
  int i = blockIdx.x * blockDim.x + threadIdx.x;
  if (i < n) out[i] = 0.0f;
}

// Convert W[N=ntiles*16][K] (fp32, row-major) into bf16 B-fragments:
// layout [kstep][ntile][lane 0..31][16 bf16], matching 32x16 16-bit B layout:
// lanes 0-15: N=lane, K=kb..kb+7,kb+16..kb+23 ; lanes 16-31: N=lane-16, K=kb+8..kb+15,kb+24..kb+31.
__global__ void swizzle_w_kernel(const float* __restrict__ W,
                                 unsigned short* __restrict__ outw, int K) {
  int idx = blockIdx.x * blockDim.x + threadIdx.x;
  int nfrags = (K / 32) * 16;
  if (idx >= nfrags * 32) return;
  int lane = idx & 31;
  int f = idx >> 5;
  int ks = f >> 4;
  int nt = f & 15;
  int n  = nt * 16 + (lane & 15);
  int kb = ks * 32 + ((lane & 16) ? 8 : 0);
  const float* src = W + (size_t)n * K + kb;
  unsigned int pk[8];
#pragma unroll
  for (int p = 0; p < 8; ++p)
    pk[p] = (unsigned)f2bf(src[2 * p + (p >> 2) * 8]) | ((unsigned)f2bf(src[2 * p + (p >> 2) * 8 + 1]) << 16);
  vu4* dst = reinterpret_cast<vu4*>(outw + (size_t)f * 512 + lane * 16);
  vu4 a = {pk[0], pk[1], pk[2], pk[3]};
  vu4 b = {pk[4], pk[5], pk[6], pk[7]};
  dst[0] = a;
  dst[1] = b;
}

__global__ __launch_bounds__(256) void psmodel_main_kernel(
    const float* __restrict__ ps, const int* __restrict__ numbers,
    const int* __restrict__ batch, const float* __restrict__ W_comp,
    const float* __restrict__ W_psl, const float* __restrict__ W_out,
    const unsigned short* __restrict__ B1, const unsigned short* __restrict__ B2,
    float* __restrict__ out) {
  __shared__ unsigned short a_tile[2][TILE_M * PITCH_A];  // double-buffered stage
  __shared__ unsigned short h1_buf[TILE_M * PITCH_H];     // 64x256 bf16 (padded)
  __shared__ float psum[TILE_M];
  __shared__ float wout_s[H2_DIM];

  const int tid  = threadIdx.x;
  const int lane = tid & 31;
  const int wave = tid >> 5;      // 0..7
  const int mt   = wave >> 1;     // M-tile 0..3 (16 atoms each)
  const int nh   = wave & 1;      // N half: cols [nh*128, nh*128+128)
  const int half = lane >> 4;
  const int ln   = lane & 15;

  const int atom0 = blockIdx.x * TILE_M;

  if (tid < TILE_M) psum[tid] = 0.0f;
  wout_s[tid] = W_out[tid];       // blockDim == H2_DIM == 256

  // streaming-stage assignment: 4 threads per atom, 8 consecutive K each
  const int sa = tid >> 2;
  const int sk = (tid & 3) * 8;
  const float* psrow = ps + (size_t)(atom0 + sa) * N_FEAT + sk;

  v8f zero = {};
  v8f acc[8];
#pragma unroll
  for (int t = 0; t < 8; ++t) acc[t] = zero;

  float psl_part = 0.0f;

  // stage one 64x32 fp32 slab -> bf16 LDS buffer, accumulate psl partial
  auto stage = [&](int ks, unsigned short* buf) {
    const vf4* src = reinterpret_cast<const vf4*>(psrow + ks * 32);
    vf4 x0 = __builtin_nontemporal_load(src);       // stream ps, don't pollute caches
    vf4 x1 = __builtin_nontemporal_load(src + 1);
    const vf4* wp = reinterpret_cast<const vf4*>(W_psl + ks * 32 + sk);
    vf4 w0 = wp[0], w1 = wp[1];
    psl_part += x0.x * w0.x + x0.y * w0.y + x0.z * w0.z + x0.w * w0.w +
                x1.x * w1.x + x1.y * w1.y + x1.z * w1.z + x1.w * w1.w;
    unsigned int* dst = reinterpret_cast<unsigned int*>(&buf[sa * PITCH_A + sk]);
    dst[0] = (unsigned)f2bf(x0.x) | ((unsigned)f2bf(x0.y) << 16);
    dst[1] = (unsigned)f2bf(x0.z) | ((unsigned)f2bf(x0.w) << 16);
    dst[2] = (unsigned)f2bf(x1.x) | ((unsigned)f2bf(x1.y) << 16);
    dst[3] = (unsigned)f2bf(x1.z) | ((unsigned)f2bf(x1.w) << 16);
  };

  // ---------------- layer 1: X[64,1024] @ W_h1^T -> H1[64,256] -------------
  stage(0, a_tile[0]);
  __syncthreads();
  for (int ks = 0; ks < KSTEPS1; ++ks) {
    // prefetch + convert next slab into the other buffer while WMMAs run
    if (ks + 1 < KSTEPS1) stage(ks + 1, a_tile[(ks + 1) & 1]);

    v16bf a = gather_a(&a_tile[ks & 1][(mt * 16 + ln) * PITCH_A], half);
#pragma unroll
    for (int t = 0; t < 8; ++t) {
      v16bf b = load_b(B1, ks * 16 + nh * 8 + t, lane);
      acc[t] = __builtin_amdgcn_wmma_f32_16x16x32_bf16(
          false, a, false, b, (short)0, acc[t], false, false);
    }
    __syncthreads();   // next buffer fully staged; current reads complete
  }

  __hip_atomic_fetch_add(&psum[sa], psl_part, __ATOMIC_RELAXED,
                         __HIP_MEMORY_SCOPE_WORKGROUP);

  // SiLU + write H1 (bf16) to LDS
#pragma unroll
  for (int t = 0; t < 8; ++t) {
    int n = nh * 128 + t * 16 + ln;
#pragma unroll
    for (int v = 0; v < 8; ++v) {
      int m = mt * 16 + half * 8 + v;            // C/D layout: VGPR v, lane half
      h1_buf[m * PITCH_H + n] = f2bf(silu(acc[t][v]));
    }
  }
  __syncthreads();

  // ---------------- layer 2: H1[64,256] @ W_h2^T -> H2[64,256] -------------
#pragma unroll
  for (int t = 0; t < 8; ++t) acc[t] = zero;
  for (int ks = 0; ks < KSTEPS2; ++ks) {
    v16bf a = gather_a(&h1_buf[(mt * 16 + ln) * PITCH_H + ks * 32], half);
#pragma unroll
    for (int t = 0; t < 8; ++t) {
      v16bf b = load_b(B2, ks * 16 + nh * 8 + t, lane);
      acc[t] = __builtin_amdgcn_wmma_f32_16x16x32_bf16(
          false, a, false, b, (short)0, acc[t], false, false);
    }
  }

  // ---------------- psnn = silu(H2) . W_out, reduce per atom ---------------
  float pacc[8];
#pragma unroll
  for (int v = 0; v < 8; ++v) pacc[v] = 0.0f;
#pragma unroll
  for (int t = 0; t < 8; ++t) {
    float w = wout_s[nh * 128 + t * 16 + ln];
#pragma unroll
    for (int v = 0; v < 8; ++v) pacc[v] += silu(acc[t][v]) * w;
  }
#pragma unroll
  for (int v = 0; v < 8; ++v) {
    int m = mt * 16 + half * 8 + v;
    __hip_atomic_fetch_add(&psum[m], pacc[v], __ATOMIC_RELAXED,
                           __HIP_MEMORY_SCOPE_WORKGROUP);
  }
  __syncthreads();

  // ---------------- composition term + segment sum into out ----------------
  if (tid < TILE_M) {
    int atom = atom0 + tid;
    float comp = W_comp[numbers[atom]];          // W_comp is [1,4]
    float e = comp + 1.0f /* SCALE */ * psum[tid];
    __hip_atomic_fetch_add(&out[batch[atom]], e, __ATOMIC_RELAXED,
                           __HIP_MEMORY_SCOPE_AGENT);
  }
}

extern "C" void kernel_launch(void* const* d_in, const int* in_sizes, int n_in,
                              void* d_out, int out_size, void* d_ws, size_t ws_size,
                              hipStream_t stream) {
  const float* ps      = (const float*)d_in[0];
  const int*   numbers = (const int*)d_in[1];
  const int*   batch   = (const int*)d_in[2];
  const float* W_comp  = (const float*)d_in[3];
  const float* W_psl   = (const float*)d_in[4];
  const float* W_h1    = (const float*)d_in[5];
  const float* W_h2    = (const float*)d_in[6];
  const float* W_out   = (const float*)d_in[7];
  float* out = (float*)d_out;

  const int n_atoms  = in_sizes[1];   // 200000
  const int n_struct = out_size;      // 2000

  // workspace: swizzled bf16 B fragments
  // B1: (1024/32 ksteps)*(16 ntiles)*1KB = 512 KB ; B2: 8*16*1KB = 128 KB
  unsigned short* B1 = (unsigned short*)d_ws;
  unsigned short* B2 = (unsigned short*)((char*)d_ws + (size_t)KSTEPS1 * 16 * 1024);

  zero_out_kernel<<<(n_struct + 255) / 256, 256, 0, stream>>>(out, n_struct);

  // W_h1: (1024/32)*16 frags * 32 lanes = 16384 threads
  swizzle_w_kernel<<<16384 / 256, 256, 0, stream>>>(W_h1, B1, N_FEAT);
  // W_h2: (256/32)*16 frags * 32 lanes = 4096 threads
  swizzle_w_kernel<<<4096 / 256, 256, 0, stream>>>(W_h2, B2, H1_DIM);

  const int tiles = n_atoms / TILE_M;   // 200000/64 = 3125 exact, full EXEC tiles
  psmodel_main_kernel<<<tiles, 256, 0, stream>>>(ps, numbers, batch, W_comp,
                                                 W_psl, W_out, B1, B2, out);
}
